// CausalSelfAttention_2619930050737
// MI455X (gfx1250) — matrix operand order repaired and checked by reading, not compiled, not run on hardware
//
#include <hip/hip_runtime.h>

// ---------------------------------------------------------------------------
// CausalSelfAttention for MI455X (gfx1250, wave32, WMMA).
// bf16 WMMA (v_wmma_f32_16x16x32_bf16) for all three matmul stages.
// LDS staging via CDNA5 async DMA: global_load_async_to_lds_b128 (ASYNCcnt),
// double-buffered in the GEMMs.
// ---------------------------------------------------------------------------

#define D_MODEL 1024
#define N_HEAD  16
#define D_HEAD  64
#define BATCH   4
#define SEQ     2048
#define MROWS   (BATCH * SEQ)   // 8192
#define QKV_N   (3 * D_MODEL)   // 3072

typedef __attribute__((ext_vector_type(16))) __bf16 v16bf;
typedef __attribute__((ext_vector_type(8)))  float  v8f;
typedef __attribute__((ext_vector_type(4)))  unsigned int u32x4;

union FragU { u32x4 u[2]; v16bf v; };

// Load a 16x32 bf16 fragment from two contiguous 16-byte runs
// -> two ds_load_b128 per lane.
static __device__ inline v16bf load_frag16(const __bf16* p0, const __bf16* p1) {
  FragU f;
  f.u[0] = *(const u32x4*)p0;
  f.u[1] = *(const u32x4*)p1;
  return f.v;
}

// Async DMA of 16 bytes/lane into LDS (GVS mode: SGPR base + VGPR offset).
// LDS flat addresses carry the LDS byte offset in bits [31:0].
static __device__ inline void async_copy16(const void* lds_dst,
                                           const __bf16* gbase,
                                           unsigned byte_off) {
  unsigned laddr = (unsigned)(size_t)lds_dst;
  asm volatile("global_load_async_to_lds_b128 %0, %1, %2"
               :: "v"(laddr), "v"(byte_off), "s"(gbase)
               : "memory");
}
static __device__ inline void wait_async_le4() {
  asm volatile("s_wait_asynccnt 4" ::: "memory");
}
static __device__ inline void wait_async_0() {
  asm volatile("s_wait_asynccnt 0" ::: "memory");
}

static __device__ inline float rowmax16(float x) {
  x = fmaxf(x, __shfl_xor(x, 1, 32));
  x = fmaxf(x, __shfl_xor(x, 2, 32));
  x = fmaxf(x, __shfl_xor(x, 4, 32));
  x = fmaxf(x, __shfl_xor(x, 8, 32));
  return x;
}
static __device__ inline float rowsum16(float x) {
  x += __shfl_xor(x, 1, 32);
  x += __shfl_xor(x, 2, 32);
  x += __shfl_xor(x, 4, 32);
  x += __shfl_xor(x, 8, 32);
  return x;
}

// ---------------------------------------------------------------------------
// Elementwise f32 -> bf16
// ---------------------------------------------------------------------------
__global__ void cvt_f32_to_bf16(const float* __restrict__ in,
                                __bf16* __restrict__ out, int n) {
  int i = blockIdx.x * blockDim.x + threadIdx.x;
  for (; i < n; i += gridDim.x * blockDim.x) out[i] = (__bf16)in[i];
}

// w [K][N] f32 (row-major) -> wt [N][K] bf16 (row-major), i.e. transpose.
__global__ void cvt_transpose_bf16(const float* __restrict__ w,
                                   __bf16* __restrict__ wt, int K, int N) {
  int i = blockIdx.x * blockDim.x + threadIdx.x;
  int total = K * N;
  for (; i < total; i += gridDim.x * blockDim.x) {
    int k = i / N;
    int n = i - k * N;
    wt[(size_t)n * K + k] = (__bf16)w[i];
  }
}

// ---------------------------------------------------------------------------
// Tiled bf16 WMMA GEMM:  C[M][N] = A[M][K] * Bt[N][K]^T
// Block: 256 threads (8 waves). Tile 128x128, K-step 32, double-buffered LDS
// filled by async DMA (2 async instructions per wave per 128x32 slab).
// Each wave: 32 (M) x 64 (N) = 2x4 grid of 16x16 WMMA tiles.
// IS_QKV: scatter bf16 results into [3][B][H][T][D_HEAD]; else f32 row-major.
// ---------------------------------------------------------------------------
template <bool IS_QKV>
__global__ __launch_bounds__(256) void gemm_bf16_wmma(
    const __bf16* __restrict__ A,   // [M][K]
    const __bf16* __restrict__ Bt,  // [N][K]
    void* __restrict__ outp,
    int M, int N, int K) {
  __shared__ __align__(16) __bf16 Alds[2][128][40];  // pad 32->40 (bank spread)
  __shared__ __align__(16) __bf16 Blds[2][128][40];

  const int t  = threadIdx.x;
  const int w  = t >> 5;
  const int l  = t & 31;
  const int li = l & 15;
  const int hi = l >> 4;
  const int wm = w & 3;   // 4 m-slots of 32 rows
  const int wn = w >> 2;  // 2 n-slots of 64 cols
  const int m0 = blockIdx.x * 128;
  const int n0 = blockIdx.y * 128;

  // Stage one 128x32 slab (512 16B chunks): thread handles chunks t, t+256.
  auto stage = [&](int buf, int row0, const __bf16* __restrict__ G, int k0,
                   __bf16 (*lds)[128][40]) {
#pragma unroll
    for (int i = 0; i < 2; ++i) {
      int c  = t + i * 256;
      int m  = c >> 2;
      int kk = (c & 3) * 8;
      unsigned off =
          (unsigned)(((size_t)(row0 + m) * K + k0 + kk) * sizeof(__bf16));
      async_copy16(&lds[buf][m][kk], G, off);
    }
  };

  v8f acc[2][4] = {};
  const int NSTAGE = K >> 5;  // K/32

  // Prologue: fill stage 0 (and 1 if present).
  stage(0, m0, A, 0, Alds);
  stage(0, n0, Bt, 0, Blds);
  if (NSTAGE > 1) {
    stage(1, m0, A, 32, Alds);
    stage(1, n0, Bt, 32, Blds);
  }

  for (int s = 0; s < NSTAGE; ++s) {
    const int buf = s & 1;
    // Wait for this wave's stage-s DMAs (leave stage s+1's 4 in flight).
    if (s + 1 < NSTAGE) wait_async_le4();
    else                wait_async_0();
    __syncthreads();  // whole tile resident

    v16bf afr[2], bfr[4];
#pragma unroll
    for (int i = 0; i < 2; ++i) {
      int m = wm * 32 + i * 16 + li;
      // A 16-bit 16x32 layout: row M=li; K = hi*8+e (e<8), 16+hi*8+(e-8).
      afr[i] = load_frag16(&Alds[buf][m][hi * 8], &Alds[buf][m][16 + hi * 8]);
    }
#pragma unroll
    for (int j = 0; j < 4; ++j) {
      int n = wn * 64 + j * 16 + li;
      // B 16-bit 32x16 layout: column N=li; K = hi*8+e (e<8), 16+hi*8+(e-8).
      bfr[j] = load_frag16(&Blds[buf][n][hi * 8], &Blds[buf][n][16 + hi * 8]);
    }
#pragma unroll
    for (int i = 0; i < 2; ++i)
#pragma unroll
      for (int j = 0; j < 4; ++j)
        acc[i][j] = __builtin_amdgcn_wmma_f32_16x16x32_bf16(
            false, afr[i], false, bfr[j], (short)0, acc[i][j], false, false);

    __syncthreads();  // everyone done reading buf before DMA overwrites it
    if (s + 2 < NSTAGE) {
      stage(buf, m0, A, (s + 2) * 32, Alds);
      stage(buf, n0, Bt, (s + 2) * 32, Blds);
    }
  }

  // Epilogue. C/D layout: row = v + hi*8, col = li (per 16x16 tile).
#pragma unroll
  for (int i = 0; i < 2; ++i)
#pragma unroll
    for (int j = 0; j < 4; ++j)
#pragma unroll
      for (int v = 0; v < 8; ++v) {
        int r = m0 + wm * 32 + i * 16 + v + hi * 8;
        int c = n0 + wn * 64 + j * 16 + li;
        float val = acc[i][j][v];
        if (IS_QKV) {
          __bf16* qkv = (__bf16*)outp;
          int which = c >> 10;      // 0=q 1=k 2=v
          int cc    = c & 1023;
          int head  = cc >> 6;
          int dh    = cc & 63;
          int b     = r >> 11;      // r = b*SEQ + t
          int tt    = r & 2047;
          size_t off =
              ((((size_t)which * BATCH + b) * N_HEAD + head) * SEQ + tt) * D_HEAD + dh;
          qkv[off] = (__bf16)val;
        } else {
          ((float*)outp)[(size_t)r * N + c] = val;
        }
      }
}

// ---------------------------------------------------------------------------
// Flash attention: one block (128 thr = 4 waves) per 64 query rows per (b,h).
// Each wave owns 16 query rows. Streams 64-key blocks through LDS with
// online softmax; causal mask on diagonal blocks. K tiles staged by async
// DMA; V transposed through the VALU path (layout change in flight).
// ---------------------------------------------------------------------------
__global__ __launch_bounds__(128) void attn_fwd(const __bf16* __restrict__ qkv,
                                                __bf16* __restrict__ y) {
  __shared__ __align__(16) __bf16 Klds[64][72];     // [key][dh]
  __shared__ __align__(16) __bf16 Vtld[64][72];     // [dh][key] (transposed)
  __shared__ __align__(16) __bf16 Plds[4][16][72];  // per-wave P tile

  const int t  = threadIdx.x;
  const int w  = t >> 5;
  const int l  = t & 31;
  const int li = l & 15;
  const int hi = l >> 4;
  const int q0 = blockIdx.x * 64;
  const int h  = blockIdx.y;
  const int b  = blockIdx.z;

  const size_t plane   = (size_t)BATCH * N_HEAD * SEQ * D_HEAD;
  const size_t headoff = (((size_t)b * N_HEAD) + h) * (size_t)SEQ * D_HEAD;
  const __bf16* Qm = qkv + 0 * plane + headoff;
  const __bf16* Km = qkv + 1 * plane + headoff;
  const __bf16* Vm = qkv + 2 * plane + headoff;

  // Q fragments for this wave's 16 rows (A-layout, Dh=64 -> 2 frags of K=32).
  v16bf qf[2];
  {
    int qrow = q0 + w * 16 + li;
#pragma unroll
    for (int ks = 0; ks < 2; ++ks)
      qf[ks] = load_frag16(&Qm[(size_t)qrow * D_HEAD + ks * 32 + hi * 8],
                           &Qm[(size_t)qrow * D_HEAD + ks * 32 + 16 + hi * 8]);
  }

  float rmax[8], rsum[8];
  v8f Oacc[4] = {};
#pragma unroll
  for (int v = 0; v < 8; ++v) { rmax[v] = -1e30f; rsum[v] = 0.0f; }

  for (int j0 = 0; j0 <= q0 + 63; j0 += 64) {   // uniform across the block
    __syncthreads();   // previous tile fully consumed before restaging
    // Stage K block [64 keys][64 dh] via async DMA (4 instr / wave).
#pragma unroll
    for (int i = 0; i < 4; ++i) {
      int vi  = t + i * 128;
      int key = vi >> 3;
      int dd  = (vi & 7) * 8;
      unsigned off =
          (unsigned)(((size_t)(j0 + key) * D_HEAD + dd) * sizeof(__bf16));
      async_copy16(&Klds[key][dd], Km, off);
    }
    // Stage V transposed: Vt[dh][key].
#pragma unroll
    for (int i = 0; i < 32; ++i) {
      int idx = t + i * 128;
      int key = idx >> 6;
      int dh  = idx & 63;
      Vtld[dh][key] = Vm[(size_t)(j0 + key) * D_HEAD + dh];
    }
    wait_async_0();
    __syncthreads();

    // S = Q * K^T for 4 key tiles of 16.
    v8f S[4] = {};
#pragma unroll
    for (int jt = 0; jt < 4; ++jt) {
      int key = jt * 16 + li;
#pragma unroll
      for (int ks = 0; ks < 2; ++ks) {
        v16bf kf = load_frag16(&Klds[key][ks * 32 + hi * 8],
                               &Klds[key][ks * 32 + 16 + hi * 8]);
        S[jt] = __builtin_amdgcn_wmma_f32_16x16x32_bf16(
            false, qf[ks], false, kf, (short)0, S[jt], false, false);
      }
    }

    // Scale (1/sqrt(64)) + causal mask.
#pragma unroll
    for (int jt = 0; jt < 4; ++jt)
#pragma unroll
      for (int v = 0; v < 8; ++v) {
        float s  = S[jt][v] * 0.125f;
        int   jg = j0 + jt * 16 + li;
        int   qg = q0 + w * 16 + v + hi * 8;
        S[jt][v] = (jg <= qg) ? s : -1e30f;
      }

    // Online softmax update (per row = v + hi*8).
#pragma unroll
    for (int v = 0; v < 8; ++v) {
      float nm = fmaxf(fmaxf(S[0][v], S[1][v]), fmaxf(S[2][v], S[3][v]));
      nm = rowmax16(nm);
      nm = fmaxf(rmax[v], nm);
      float alpha = __expf(rmax[v] - nm);
      rmax[v] = nm;
      float ps = 0.0f;
#pragma unroll
      for (int jt = 0; jt < 4; ++jt) {
        float p = __expf(S[jt][v] - nm);
        ps += p;
        Plds[w][v + hi * 8][jt * 16 + li] = (__bf16)p;  // C-layout -> LDS
      }
      ps = rowsum16(ps);
      rsum[v] = rsum[v] * alpha + ps;
#pragma unroll
      for (int n = 0; n < 4; ++n) Oacc[n][v] = Oacc[n][v] * alpha;
    }
    __syncthreads();

    // O += P * V  (P reloaded from LDS in A-layout; V from transposed LDS).
    v16bf pf[2];
#pragma unroll
    for (int ks = 0; ks < 2; ++ks)
      pf[ks] = load_frag16(&Plds[w][li][ks * 32 + hi * 8],
                           &Plds[w][li][ks * 32 + 16 + hi * 8]);
#pragma unroll
    for (int n = 0; n < 4; ++n)
#pragma unroll
      for (int ks = 0; ks < 2; ++ks) {
        v16bf vf = load_frag16(&Vtld[n * 16 + li][ks * 32 + hi * 8],
                               &Vtld[n * 16 + li][ks * 32 + 16 + hi * 8]);
        Oacc[n] = __builtin_amdgcn_wmma_f32_16x16x32_bf16(
            false, pf[ks], false, vf, (short)0, Oacc[n], false, false);
      }
  }

  // Normalize and write y[b][t][h*64+dh] (bf16, feeds the projection GEMM).
#pragma unroll
  for (int n = 0; n < 4; ++n)
#pragma unroll
    for (int v = 0; v < 8; ++v) {
      int qg = q0 + w * 16 + v + hi * 8;
      int dh = n * 16 + li;
      y[((size_t)b * SEQ + qg) * D_MODEL + h * D_HEAD + dh] =
          (__bf16)(Oacc[n][v] / rsum[v]);
    }
}

// ---------------------------------------------------------------------------
// Host-side launcher
// ---------------------------------------------------------------------------
extern "C" void kernel_launch(void* const* d_in, const int* in_sizes, int n_in,
                              void* d_out, int out_size, void* d_ws,
                              size_t ws_size, hipStream_t stream) {
  (void)in_sizes; (void)n_in; (void)out_size; (void)ws_size;
  const float* x     = (const float*)d_in[0];  // [B,T,C]
  const float* wqkv  = (const float*)d_in[1];  // [C,3C]
  const float* wproj = (const float*)d_in[2];  // [C,C]
  float* out = (float*)d_out;                  // [B,T,C] f32

  char* ws = (char*)d_ws;
  // Workspace layout (bytes):
  //   xb     @ 0      : 16 MiB   x in bf16
  //   wqkvT  @ 16 MiB :  6 MiB   w_qkv^T in bf16 [3072][1024]
  //   wprojT @ 22 MiB :  2 MiB   w_proj^T in bf16 [1024][1024]
  //   qkvbuf @ 24 MiB : 48 MiB   [3][B][H][T][64] bf16
  //   yb     @ 72 MiB : 16 MiB   attention output bf16  (total 88 MiB)
  __bf16* xb     = (__bf16*)(ws);
  __bf16* wqkvT  = (__bf16*)(ws + (size_t)16 * 1024 * 1024);
  __bf16* wprojT = (__bf16*)(ws + (size_t)22 * 1024 * 1024);
  __bf16* qkvbuf = (__bf16*)(ws + (size_t)24 * 1024 * 1024);
  __bf16* yb     = (__bf16*)(ws + (size_t)72 * 1024 * 1024);

  // 1) Precision conversion / weight transposition.
  cvt_f32_to_bf16<<<2048, 256, 0, stream>>>(x, xb, MROWS * D_MODEL);
  cvt_transpose_bf16<<<(D_MODEL * QKV_N + 255) / 256, 256, 0, stream>>>(
      wqkv, wqkvT, D_MODEL, QKV_N);
  cvt_transpose_bf16<<<(D_MODEL * D_MODEL + 255) / 256, 256, 0, stream>>>(
      wproj, wprojT, D_MODEL, D_MODEL);

  // 2) QKV projection, scattered into [3][B][H][T][64] bf16.
  gemm_bf16_wmma<true><<<dim3(MROWS / 128, QKV_N / 128), 256, 0, stream>>>(
      xb, wqkvT, qkvbuf, MROWS, QKV_N, D_MODEL);

  // 3) Causal flash attention -> yb [B,T,C] bf16.
  attn_fwd<<<dim3(SEQ / 64, N_HEAD, BATCH), 128, 0, stream>>>(qkvbuf, yb);

  // 4) Output projection -> f32 result.
  gemm_bf16_wmma<false><<<dim3(MROWS / 128, D_MODEL / 128), 256, 0, stream>>>(
      yb, wprojT, out, MROWS, D_MODEL, D_MODEL);
}
